// CLSControlledDynamicBlock_7387343749867
// MI455X (gfx1250) — compile-verified, matrix operand order checked by reading, not run on hardware
//
#include <hip/hip_runtime.h>
#include <cmath>

typedef __attribute__((ext_vector_type(16))) _Float16     v16h;
typedef __attribute__((ext_vector_type(8)))  _Float16     v8h;
typedef __attribute__((ext_vector_type(8)))  float        v8f;
typedef __attribute__((ext_vector_type(4)))  unsigned int v4u;
typedef __attribute__((ext_vector_type(8)))  int          v8i;
typedef __attribute__((ext_vector_type(4)))  int          v4i;

#define B_      16
#define EMB_    768
#define HID_    192
#define CIN_    64
#define COUT_   64
#define H_      112
#define W_      112
#define HW_     (H_ * W_)                 // 12544
#define TOTAL_  (COUT_ * CIN_ * 9)        // 36864
#define RED_    (CIN_ * 9)                // 576
#define NCHUNK_ 18                        // 576 / 32 K-slices
#define NTILE_  4                         // 64 couts / 16

// workspace layout (bytes):
//   [0, 12288)          : h (16 x 192 f32)
//   [16384, ~1.2M)      : packed WMMA B-fragments (f16)
//                         [b][chunk 18][ntile 4][lane 32][16 halves]
//   [2M, 2M+25.7M)      : featT (f16)  [b][pixel 12544][cin 64]
#define KPACK_OFF 16384
#define FEATT_OFF (2u * 1024u * 1024u)

#define LDSB_HALVES (9 * NTILE_ * 32 * 16)   // one 9-chunk phase = 36 KB
#define TDM_QWORDS  (LDSB_HALVES / 4)        // 4608 8-byte elements

// ---------------------------------------------------------------- stage 1
__global__ void mlp_h_kernel(const float* __restrict__ cls,
                             const float* __restrict__ W1,
                             const float* __restrict__ b1,
                             float* __restrict__ h) {
    __shared__ float xs[B_ * EMB_];
    for (int i = threadIdx.x; i < B_ * EMB_; i += blockDim.x) xs[i] = cls[i];
    __syncthreads();
    const int j = threadIdx.x;
    if (j < HID_) {
        float acc[B_];
#pragma unroll
        for (int b = 0; b < B_; ++b) acc[b] = 0.f;
        for (int k = 0; k < EMB_; ++k) {
            const float w = W1[k * HID_ + j];
#pragma unroll
            for (int b = 0; b < B_; ++b) acc[b] = fmaf(xs[b * EMB_ + k], w, acc[b]);
        }
        const float bb = b1[j];
#pragma unroll
        for (int b = 0; b < B_; ++b) h[b * HID_ + j] = fmaxf(acc[b] + bb, 0.f);
    }
}

// ---------------------------------------------------------------- stage 2
// params = tanh(h @ W2 + b2), written directly into WMMA B-fragment layout.
// Reduction index kk = tap*64 + cin (tap-major: each 32-slice is half a tap).
__global__ void mlp_params_kernel(const float* __restrict__ h,
                                  const float* __restrict__ W2,
                                  const float* __restrict__ b2,
                                  _Float16* __restrict__ kpack) {
    __shared__ float hs[B_ * HID_];
    for (int i = threadIdx.x; i < B_ * HID_; i += blockDim.x) hs[i] = h[i];
    __syncthreads();

    const int j = blockIdx.x * blockDim.x + threadIdx.x;   // [0, 36864)
    float acc[B_];
#pragma unroll
    for (int b = 0; b < B_; ++b) acc[b] = 0.f;
    for (int k = 0; k < HID_; ++k) {
        const float w = W2[(size_t)k * TOTAL_ + j];        // coalesced over j
#pragma unroll
        for (int b = 0; b < B_; ++b) acc[b] = fmaf(hs[b * HID_ + k], w, acc[b]);
    }
    const float bias = b2[j];

    const int cout  = j / RED_;
    const int r     = j % RED_;
    const int cin   = r / 9;
    const int tap   = r % 9;
    const int kk    = tap * CIN_ + cin;
    const int chunk = kk >> 5;
    const int kloc  = kk & 31;
    const int ntile = cout >> 4;
    const int n     = cout & 15;
    const int hi    = (kloc >> 3) & 1;
    const int lane  = n + (hi << 4);
    const int idx16 = (kloc & 7) | ((kloc & 16) >> 1);

#pragma unroll
    for (int b = 0; b < B_; ++b) {
        const float v = tanhf(acc[b] + bias);
        const size_t off =
            ((((size_t)b * NCHUNK_ + chunk) * NTILE_ + ntile) * 32 + lane) * 16 + idx16;
        kpack[off] = (_Float16)v;
    }
}

// ---------------------------------------------------------------- stage 2.5
// Transpose + convert: featT[b][p][cin] = (f16) features[b][cin][p].
__global__ void __launch_bounds__(256)
transpose_f16_kernel(const float* __restrict__ feat,
                     _Float16* __restrict__ featT) {
    const int p = blockIdx.x * 256 + threadIdx.x;     // [0, 12544)
    const int b = blockIdx.y;
    const float* fp = feat + (size_t)b * CIN_ * HW_ + p;
    _Float16 row[CIN_];
#pragma unroll
    for (int c = 0; c < CIN_; ++c) row[c] = (_Float16)fp[(size_t)c * HW_];
    v8h* dst = (v8h*)(featT + ((size_t)b * HW_ + p) * CIN_);
#pragma unroll
    for (int i = 0; i < 8; ++i) dst[i] = *(const v8h*)(row + i * 8);
}

// ---------------------------------------------------------------- stage 3
// Implicit-GEMM conv via v_wmma_f32_16x16x32_f16.
// B-fragment slabs (36 KB/phase) are DMA'd into LDS by the Tensor Data Mover:
// wave 0 issues tensor_load_to_lds with a 1-D D# (data_size=8B, dim0=4608),
// waits on TENSORcnt, then the workgroup barrier publishes LDS to all waves.
__global__ void __launch_bounds__(256)
conv_wmma_kernel(const float* __restrict__ feat,
                 const _Float16* __restrict__ featT,
                 const _Float16* __restrict__ kpack,
                 float* __restrict__ out) {
    __shared__ __align__(16) _Float16 bsh[LDSB_HALVES];

    const int lane  = threadIdx.x & 31;
    const int wave  = threadIdx.x >> 5;
    const int ptile = blockIdx.x * 8 + wave;   // [0, 784); tiles never cross a row
    const int b     = blockIdx.y;
    const int hi    = lane >> 4;

    const int m  = lane & 15;
    const int p0 = ptile * 16 + m;
    const int y  = p0 / W_;
    const int x  = p0 % W_;

    v8f acc[NTILE_] = {};

    const _Float16* kb  = kpack + (size_t)b * NCHUNK_ * NTILE_ * 32 * 16;
    const _Float16* ftb = featT + (size_t)b * HW_ * CIN_;

    // LDS byte address of the staging slab (generic LDS ptr = {aperture, offset})
    const unsigned int lds_addr = (unsigned int)(size_t)&bsh[0];

    for (int phase = 0; phase < 2; ++phase) {
        __syncthreads();                      // previous phase fully consumed
        if (threadIdx.x < 32) {               // wave 0 drives the TDM
            const unsigned long long ga =
                (unsigned long long)(size_t)(kb + (size_t)phase * LDSB_HALVES);
            // D# group 0: count=1 | lds_addr | global_addr[56:0] | type=2
            const v4u g0 = {1u, lds_addr, (unsigned int)ga,
                            (unsigned int)((ga >> 32) & 0x1FFFFFFu) | 0x80000000u};
            // D# group 1: data_size=3 (8B); tensor_dim0=tile_dim0=4608;
            // tensor_dim1=tile_dim1=1; tensor_dim0_stride=4608
            const v8i g1 = {0x00030000,                      // wg_mask=0, ds=8B
                            (int)((TDM_QWORDS & 0xFFFF) << 16),   // dim0 lo
                            0x00010000,                      // dim0 hi | dim1=1
                            (int)((TDM_QWORDS & 0xFFFF) << 16),   // tile_dim0
                            1,                               // tile_dim1=1
                            TDM_QWORDS,                      // dim0_stride lo
                            0, 0};
            const v4i gz4 = {0, 0, 0, 0};
#if __clang_major__ >= 23
            const v8i gz8 = {0, 0, 0, 0, 0, 0, 0, 0};
            __builtin_amdgcn_tensor_load_to_lds(g0, g1, gz4, gz4, gz8, 0);
#else
            __builtin_amdgcn_tensor_load_to_lds(g0, g1, gz4, gz4, 0);
#endif
            __builtin_amdgcn_s_wait_tensorcnt(0);
        }
        __syncthreads();                      // LDS slab visible to all waves

#pragma unroll
        for (int cl = 0; cl < 9; ++cl) {
            const int chunk = phase * 9 + cl;
            const int tap   = chunk >> 1;
            const int dy    = tap / 3 - 1;
            const int dx    = tap % 3 - 1;
            const int cbase = (chunk & 1) * 32;
            const int yy    = y + dy;
            const int xx    = x + dx;
            const bool valid = (yy >= 0) & (yy < H_) & (xx >= 0) & (xx < W_);

            // A-fragment: two contiguous 16B f16 loads per lane (no cvt)
            const _Float16* ap =
                ftb + (size_t)(yy * W_ + xx) * CIN_ + cbase + hi * 8;
            const v8h zero = {};
            const v8h alo = valid ? *(const v8h*)(ap)      : zero;
            const v8h ahi = valid ? *(const v8h*)(ap + 16) : zero;
            const v16h a = __builtin_shufflevector(alo, ahi,
                0, 1, 2, 3, 4, 5, 6, 7, 8, 9, 10, 11, 12, 13, 14, 15);

            const _Float16* kc = bsh + (size_t)cl * NTILE_ * 32 * 16;
#pragma unroll
            for (int nt = 0; nt < NTILE_; ++nt) {
                const v16h bf = *(const v16h*)(kc + ((size_t)nt * 32 + lane) * 16);
                acc[nt] = __builtin_amdgcn_wmma_f32_16x16x32_f16(
                    false, a, false, bf, (short)0, acc[nt], false, false);
            }
        }
    }

    // D rows are 8 consecutive pixels per lane: vectorized residual + store.
    const int n = lane & 15;
#pragma unroll
    for (int nt = 0; nt < NTILE_; ++nt) {
        const int    cout = nt * 16 + n;
        const size_t idx0 =
            ((size_t)b * COUT_ + cout) * HW_ + ptile * 16 + hi * 8;
        const v8f res = *(const v8f*)(feat + idx0);
        *(v8f*)(out + idx0) = acc[nt] + res;
    }
}

extern "C" void kernel_launch(void* const* d_in, const int* in_sizes, int n_in,
                              void* d_out, int out_size, void* d_ws, size_t ws_size,
                              hipStream_t stream) {
    const float* cls  = (const float*)d_in[0];   // (16,1,768)
    const float* feat = (const float*)d_in[1];   // (16,64,112,112)
    const float* W1   = (const float*)d_in[2];   // (768,192)
    const float* b1   = (const float*)d_in[3];   // (192)
    const float* W2   = (const float*)d_in[4];   // (192,36864)
    const float* b2   = (const float*)d_in[5];   // (36864)
    float*       out  = (float*)d_out;           // (16,64,112,112)

    float*    h     = (float*)d_ws;
    _Float16* kpack = (_Float16*)((char*)d_ws + KPACK_OFF);
    _Float16* featT = (_Float16*)((char*)d_ws + FEATT_OFF);

    mlp_h_kernel<<<1, 192, 0, stream>>>(cls, W1, b1, h);
    mlp_params_kernel<<<TOTAL_ / 256, 256, 0, stream>>>(h, W2, b2, kpack);

    dim3 tgrid(HW_ / 256, B_);                    // 49 x 16
    transpose_f16_kernel<<<tgrid, 256, 0, stream>>>(feat, featT);

    dim3 cgrid(98, B_);                           // 98*8 waves = 784 ptiles
    conv_wmma_kernel<<<cgrid, 256, 0, stream>>>(feat, featT, kpack, out);
}
